// GraphAttentionalLayer_4458176053479
// MI455X (gfx1250) — matrix operand order repaired and checked by reading, not compiled, not run on hardware
//
#include <hip/hip_runtime.h>
#include <hip/hip_bf16.h>

// ---------------------------------------------------------------------------
// GAT layer, fused for CDNA5 (gfx1250, wave32, WMMA + async global->LDS).
//
// softmax(src_i + tgt_j) over masked j == softmax(tgt_j) masked (src_i is
// constant along the softmax axis and cancels).  So:
//   out[i,d] = (1/4) * sum_h (Adj @ (w_h * ht_h))[i,d] / (Adj @ w_h)[i]
// with w[h,j] = exp(tgt[h,j] - max_h).  Numerator + denominator fuse into ONE
// [8192 x 8192] x [8192 x 528] bf16 WMMA GEMM (cols 0..511 data, 512..515 w,
// 516..527 zero pad); A = adjacency converted to 0/1 bf16 while staging.
// ---------------------------------------------------------------------------

typedef __bf16 bf16_t;
typedef __attribute__((ext_vector_type(16))) __bf16 v16bf;
typedef __attribute__((ext_vector_type(8)))  __bf16 v8bf;
typedef __attribute__((ext_vector_type(8)))  float  v8f;
typedef __attribute__((ext_vector_type(4)))  float  v4f;
typedef __attribute__((ext_vector_type(4)))  int    v4i;

#if defined(__has_builtin)
#if __has_builtin(__builtin_amdgcn_sched_group_barrier)
#define SCHED_GROUP(mask, size, id) __builtin_amdgcn_sched_group_barrier((mask), (size), (id))
#endif
#endif
#ifndef SCHED_GROUP
#define SCHED_GROUP(mask, size, id)
#endif

// Async global->LDS copy (ASYNCcnt-tracked), GVS addressing:
//   each active lane copies 16 B from saddr + voff to LDS at its lds_off
__device__ __forceinline__ void async_b128_to_lds(unsigned lds_off, unsigned voff,
                                                  const void* sbase)
{
    asm volatile("global_load_async_to_lds_b128 %0, %1, %2"
                 :: "v"(lds_off), "v"(voff), "s"(sbase) : "memory");
}
__device__ __forceinline__ void wait_asynccnt0()
{
    asm volatile("s_wait_asynccnt 0x0" ::: "memory");
}

#define NV      8192      // nodes
#define DIN     256
#define DH      128
#define NH      4
#define NC1     512       // cols of ht (= NH*DH)
#define NC2     528       // cols of big GEMM (512 data + 4 denom + 12 pad)

// ---- workspace layout (bytes) ---------------------------------------------
#define WT_OFF    0u                       // bf16 [512][256]   W transposed
#define WT_BYTES  (512u*256u*2u)
#define HT_OFF    (WT_OFF + WT_BYTES)      // f32  [8192][512]  leakyrelu(h@W)
#define HT_BYTES  (8192u*512u*4u)
#define TGT_OFF   (HT_OFF + HT_BYTES)      // f32  [4][8192]
#define TGT_BYTES (4u*8192u*4u)
#define HMAX_OFF  (TGT_OFF + TGT_BYTES)    // f32  [4]
#define HMAX_BYTES 256u
#define BT2_OFF   (HMAX_OFF + HMAX_BYTES)  // bf16 [528][8192]  scaled V, K-major
#define BT2_BYTES (528u*8192u*2u)
#define C2_OFF    (BT2_OFF + BT2_BYTES)    // f32  [8192][528]
#define C2_BYTES  (8192u*528u*4u)

// ---- LDS layout for staged adjacency GEMM ---------------------------------
#define SA        80                       // bytes/row of A slab (32 bf16 + pad)
#define SB        80                       // bytes/col of B slab (32 bf16 + pad)
#define A_SLAB    (64 * SA)                // 5120 B
#define B_SLAB    (NC2 * SB)               // 42240 B
#define LDS_TOTAL (2*A_SLAB + 2*B_SLAB)    // 94720 B (dynamic LDS)

#define NT2       11                       // n-tiles per wave (3 waves * 11 * 16 = 528)
#define WN2       3
#define WM2       4

// ---------------------------------------------------------------------------
// GEMM1 (small, direct-from-global):  C = leakyrelu(A_f32[M x K] * BT[N x K]^T)
// ---------------------------------------------------------------------------
template<int NT>
__global__ void gat_wmma_gemm_f32a(const float* __restrict__ A,
                                   const bf16_t* __restrict__ BT,
                                   float* __restrict__ C,
                                   int K, int lda, int ldbt, int ldc)
{
    const int lane = threadIdx.x;
    const int half = lane >> 4;
    const int l15  = lane & 15;
    const int m0   = (blockIdx.x * blockDim.z + threadIdx.z) * 16;
    const int nt0  = (blockIdx.y * blockDim.y + threadIdx.y) * NT;

    v8f acc[NT];
#pragma unroll
    for (int t = 0; t < NT; ++t)
#pragma unroll
        for (int r = 0; r < 8; ++r) acc[t][r] = 0.0f;

    const int arow = m0 + l15;
    for (int kb = 0; kb < K; kb += 32) {
        v16bf aF;
        const float* ap = A + (size_t)arow * lda + kb + half * 8;
        v4f f0 = *(const v4f*)(ap);
        v4f f1 = *(const v4f*)(ap + 4);
        v4f f2 = *(const v4f*)(ap + 16);
        v4f f3 = *(const v4f*)(ap + 20);
#pragma unroll
        for (int i = 0; i < 4; ++i) {
            aF[i] = (bf16_t)f0[i]; aF[4+i] = (bf16_t)f1[i];
            aF[8+i] = (bf16_t)f2[i]; aF[12+i] = (bf16_t)f3[i];
        }
#pragma unroll
        for (int t = 0; t < NT; ++t) {
            const bf16_t* bp = BT + (size_t)((nt0 + t) * 16 + l15) * ldbt + kb + half * 8;
            v8bf blo = *(const v8bf*)(bp);
            v8bf bhi = *(const v8bf*)(bp + 16);
            v16bf bF;
#pragma unroll
            for (int i = 0; i < 8; ++i) { bF[i] = blo[i]; bF[8+i] = bhi[i]; }
            acc[t] = __builtin_amdgcn_wmma_f32_16x16x32_bf16(
                false, aF, false, bF, (short)0, acc[t], false, false);
        }
    }
#pragma unroll
    for (int t = 0; t < NT; ++t) {
        const int c = (nt0 + t) * 16 + l15;
#pragma unroll
        for (int r = 0; r < 8; ++r) {
            float v = acc[t][r];
            v = (v >= 0.0f) ? v : 0.1f * v;                 // fused LeakyReLU
            C[(size_t)(m0 + r + half * 8) * ldc + c] = v;
        }
    }
}

// ---------------------------------------------------------------------------
// GEMM2: C2[8192 x 528] = Adj(0/1) @ V.  LDS double-buffered pipeline:
//   - B slab staged with global_load_async_to_lds_b128 (ASYNCcnt, no VGPRs)
//   - A slab staged through VGPRs (int32 -> 0/1 bf16 convert) + ds_store
//   - compute: 3-slot B window, sched_group_barrier-pinned
//       6 ds_reads -> (1 WMMA, 2 ds_reads) x 9 -> 2 WMMAs
// block = (32, WN2=3, WM2=4) = 384 threads, 64 rows x 528 cols per workgroup.
// ---------------------------------------------------------------------------
__global__ void gat_adj_gemm(const int* __restrict__ adj,
                             const bf16_t* __restrict__ BT,
                             float* __restrict__ C)
{
    extern __shared__ __align__(16) char lds[];

    const int lane = threadIdx.x;
    const int half = lane >> 4;
    const int l15  = lane & 15;
    const int wv_n = threadIdx.y;                       // 0..2
    const int wv_m = threadIdx.z;                       // 0..3
    const int tid  = threadIdx.x + 32 * (threadIdx.y + WN2 * threadIdx.z);
    const int m0wg = blockIdx.x * 64;

    // ---- A staging: 64 rows * 4 threads, 8 ints (32 B) each ----
    const bool aAct  = tid < 256;
    const int  arow  = tid >> 2;
    const int  apart = tid & 3;
    const int* agA   = adj + (size_t)(m0wg + arow) * NV + apart * 8;
    const int  aLds  = arow * SA + apart * 16;

    // ---- B staging: chunk it covers cols [it*96 + tid>>2], 16 B each.
    // idx = it*384 + tid -> col steps by exactly 96 per it, part invariant.
    // Byte offsets into BT (global) and into B slab (LDS):
    const unsigned bGOff0 = (unsigned)((tid >> 2) * (NV * 2) + (tid & 3) * 16);
    const int      bLds0  = (tid >> 2) * SB + (tid & 3) * 16;
    // it < 5 always valid (1919 < 2112); it == 5 valid iff tid < 192.
#define B_OK(it)  ((it) < 5 || tid < 192)
#define B_GB(it)  ((unsigned)(it) * (96u * NV * 2u))    // global byte stride/chunk
#define B_LB(it)  ((it) * (96 * SB))                    // LDS byte stride/chunk

    v8f acc[NT2];
#pragma unroll
    for (int t = 0; t < NT2; ++t)
#pragma unroll
        for (int r = 0; r < 8; ++r) acc[t][r] = 0.0f;

    const int nt0   = wv_n * NT2;                       // first n-tile of this wave
    const int aoff  = (wv_m * 16 + l15) * SA + half * 32;
    const int bbase = (nt0 * 16 + l15) * SB + half * 32;
    const bf16_t one = (bf16_t)1.0f, zer = (bf16_t)0.0f;

    // ---- prologue: stage k-block 0 into buffer 0 ----
    {
        if (aAct) {
            v4i a0 = *(const v4i*)(agA);
            v4i a1 = *(const v4i*)(agA + 4);
            v8bf av;
#pragma unroll
            for (int i = 0; i < 4; ++i) {
                av[i]     = (a0[i] > 0) ? one : zer;
                av[4 + i] = (a1[i] > 0) ? one : zer;
            }
            *(v8bf*)(lds + aLds) = av;
        }
#pragma unroll
        for (int it = 0; it < 6; ++it)
            if (B_OK(it))
                async_b128_to_lds((unsigned)(2 * A_SLAB + bLds0 + B_LB(it)),
                                  bGOff0 + B_GB(it), BT);
    }
    wait_asynccnt0();
    __syncthreads();

    // ---- main pipelined loop ----
    for (int kb = 0; kb < NV; kb += 32) {
        const int  cur  = (kb >> 5) & 1;
        const bool more = (kb + 32) < NV;
        const int  kn   = kb + 32;

        // 1) stage NEXT k-block: A -> VGPRs (regular loads), B -> LDS (async).
        v4i aR0, aR1;
        if (more) {
            if (aAct) {
                aR0 = *(const v4i*)(agA + kn);
                aR1 = *(const v4i*)(agA + kn + 4);
                __builtin_prefetch((const void*)(agA + kn + 1024), 0, 1);  // adj HBM stream
            }
            const unsigned nBoff = (unsigned)(2 * A_SLAB + (cur ? 0 : B_SLAB));
#pragma unroll
            for (int it = 0; it < 6; ++it)
                if (B_OK(it))
                    async_b128_to_lds(nBoff + (unsigned)(bLds0 + B_LB(it)),
                                      bGOff0 + B_GB(it) + (unsigned)(kn * 2), BT);
        }

        // 2) compute on CURRENT LDS buffers
        const char* bufA = lds + (cur ? A_SLAB : 0);
        const char* bufB = lds + 2 * A_SLAB + (cur ? B_SLAB : 0);

        v8bf alo = *(const v8bf*)(bufA + aoff);
        v8bf ahi = *(const v8bf*)(bufA + aoff + 16);
        v16bf aF;
#pragma unroll
        for (int i = 0; i < 8; ++i) { aF[i] = alo[i]; aF[8 + i] = ahi[i]; }

        // 3-slot software window: tile t+2 loads while WMMA t runs
        v16bf bW[3];
#pragma unroll
        for (int p = 0; p < 2; ++p) {
            v8bf lo = *(const v8bf*)(bufB + bbase + p * (16 * SB));
            v8bf hi = *(const v8bf*)(bufB + bbase + p * (16 * SB) + 16);
#pragma unroll
            for (int i = 0; i < 8; ++i) { bW[p][i] = lo[i]; bW[p][8 + i] = hi[i]; }
        }
#pragma unroll
        for (int t = 0; t < NT2; ++t) {
            if (t + 2 < NT2) {
                const int s = (t + 2) % 3;
                v8bf lo = *(const v8bf*)(bufB + bbase + (t + 2) * (16 * SB));
                v8bf hi = *(const v8bf*)(bufB + bbase + (t + 2) * (16 * SB) + 16);
#pragma unroll
                for (int i = 0; i < 8; ++i) { bW[s][i] = lo[i]; bW[s][8 + i] = hi[i]; }
            }
            acc[t] = __builtin_amdgcn_wmma_f32_16x16x32_bf16(
                false, aF, false, bW[t % 3], (short)0, acc[t], false, false);
        }

        // Scheduling pipeline: 6 ds_reads (A + 2 B tiles), then
        // (1 WMMA, 2 ds_reads) x 9, then 2 trailing WMMAs.
        SCHED_GROUP(0x100, 6, 0);
#pragma unroll
        for (int g = 0; g < 9; ++g) {
            SCHED_GROUP(0x008, 1, 0);
            SCHED_GROUP(0x100, 2, 0);
        }
        SCHED_GROUP(0x008, 2, 0);

        // 3) A slab: convert + store into the OTHER buffer
        if (more && aAct) {
            char* nA = lds + (cur ? 0 : A_SLAB);
            v8bf av;
#pragma unroll
            for (int i = 0; i < 4; ++i) {
                av[i]     = (aR0[i] > 0) ? one : zer;
                av[4 + i] = (aR1[i] > 0) ? one : zer;
            }
            *(v8bf*)(nA + aLds) = av;
        }
        wait_asynccnt0();          // B slab async copies complete (per wave)
        __syncthreads();
    }

    // ---- epilogue ----
#pragma unroll
    for (int t = 0; t < NT2; ++t) {
        const int c = (nt0 + t) * 16 + l15;
#pragma unroll
        for (int r = 0; r < 8; ++r)
            C[(size_t)(m0wg + wv_m * 16 + r + half * 8) * NC2 + c] = acc[t][r];
    }
#undef B_OK
#undef B_GB
#undef B_LB
}

// ---- W [256][512] f32 -> WT [512][256] bf16 --------------------------------
__global__ void gat_make_wt(const float* __restrict__ W, bf16_t* __restrict__ WT)
{
    int g = blockIdx.x * 256 + threadIdx.x;
    int c = g >> 8, k = g & 255;
    WT[g] = (bf16_t)W[(size_t)k * NC1 + c];
}

// ---- tgt[h,j] = sum_d ht[j, h*128+d] * a[h, 128+d] (one wave per (j,h)) ----
__global__ void gat_tgt(const float* __restrict__ ht, const float* __restrict__ a,
                        float* __restrict__ tgt)
{
    int wid  = (blockIdx.x * blockDim.x + threadIdx.x) >> 5;
    int lane = threadIdx.x & 31;
    int j = wid >> 2, h = wid & 3;
    float s = 0.0f;
#pragma unroll
    for (int i = 0; i < 4; ++i) {
        int d = lane + 32 * i;
        s += ht[(size_t)j * NC1 + h * DH + d] * a[h * (2 * DH) + DH + d];
    }
#pragma unroll
    for (int m = 16; m; m >>= 1) s += __shfl_xor(s, m, 32);
    if (lane == 0) tgt[h * NV + j] = s;
}

// ---- per-head max of tgt ---------------------------------------------------
__global__ void gat_head_max(const float* __restrict__ tgt, float* __restrict__ hmax)
{
    __shared__ float s[256];
    int h = blockIdx.x;
    float m = -3.0e38f;
    for (int j = threadIdx.x; j < NV; j += 256) m = fmaxf(m, tgt[h * NV + j]);
    s[threadIdx.x] = m;
    __syncthreads();
    for (int off = 128; off; off >>= 1) {
        if (threadIdx.x < off) s[threadIdx.x] = fmaxf(s[threadIdx.x], s[threadIdx.x + off]);
        __syncthreads();
    }
    if (threadIdx.x == 0) hmax[h] = s[0];
}

// ---- BT2[c][j]: cols 0..511 = w[h,j]*ht[j,c]; 512..515 = w[h,j]; rest 0 ----
__global__ void gat_build_bt(const float* __restrict__ ht, const float* __restrict__ tgt,
                             const float* __restrict__ hmax, bf16_t* __restrict__ BT2)
{
    int j = blockIdx.x * 256 + threadIdx.x;
    float w[NH];
#pragma unroll
    for (int h = 0; h < NH; ++h) w[h] = expf(tgt[h * NV + j] - hmax[h]);
    for (int c = 0; c < NC2; ++c) {
        float v;
        if (c < NC1)           v = ht[(size_t)j * NC1 + c] * w[c >> 7];
        else if (c < NC1 + NH) v = w[c - NC1];
        else                   v = 0.0f;
        BT2[(size_t)c * NV + j] = (bf16_t)v;       // coalesced along j
    }
}

// ---- out[j,d] = 0.25 * sum_h C2[j, h*128+d] / C2[j, 512+h] -----------------
__global__ void gat_finalize(const float* __restrict__ C2, float* __restrict__ out)
{
    int t = blockIdx.x * 256 + threadIdx.x;
    int j = t >> 7, d = t & 127;
    const float* row = C2 + (size_t)j * NC2;
    float s = 0.0f;
#pragma unroll
    for (int h = 0; h < NH; ++h) s += row[h * DH + d] / row[NC1 + h];
    out[t] = 0.25f * s;
}

// ---------------------------------------------------------------------------
extern "C" void kernel_launch(void* const* d_in, const int* in_sizes, int n_in,
                              void* d_out, int out_size, void* d_ws, size_t ws_size,
                              hipStream_t stream)
{
    const float* h   = (const float*)d_in[0];   // [8192][256]
    const int*   adj = (const int*)d_in[1];     // [8192][8192]
    const float* W   = (const float*)d_in[2];   // [256][512]
    const float* a   = (const float*)d_in[3];   // [4][256]
    float* out = (float*)d_out;                 // [8192][128]

    char* ws = (char*)d_ws;
    bf16_t* WT   = (bf16_t*)(ws + WT_OFF);
    float*  ht   = (float*) (ws + HT_OFF);
    float*  tgt  = (float*) (ws + TGT_OFF);
    float*  hmax = (float*) (ws + HMAX_OFF);
    bf16_t* BT2  = (bf16_t*)(ws + BT2_OFF);
    float*  C2   = (float*) (ws + C2_OFF);

    // 0) W -> WT (bf16, transposed)
    gat_make_wt<<<512, 256, 0, stream>>>(W, WT);

    // 1) ht = leakyrelu(h @ W) : WMMA, f32 A converted on the fly
    gat_wmma_gemm_f32a<8>
        <<<dim3(128, 1), dim3(32, 4, 4), 0, stream>>>(h, WT, ht, DIN, DIN, DIN, NC1);

    // 2) tgt[h,j]
    gat_tgt<<<(NV * NH) / 8, 256, 0, stream>>>(ht, a, tgt);

    // 3) per-head max
    gat_head_max<<<NH, 256, 0, stream>>>(tgt, hmax);

    // 4) scaled-value matrix (bf16, K-major)
    gat_build_bt<<<NV / 256, 256, 0, stream>>>(ht, tgt, hmax, BT2);

    // 5) big GEMM: Adj(0/1 bf16) @ V, async-staged double-buffered pipeline
    gat_adj_gemm<<<dim3(128), dim3(32, WN2, WM2), LDS_TOTAL, stream>>>(adj, BT2, C2);

    // 6) denominators + head mean
    gat_finalize<<<(NV * DH) / 256, 256, 0, stream>>>(C2, out);
}
